// POlogLSTMModel_49306224558586
// MI455X (gfx1250) — matrix operand order repaired and checked by reading, not compiled, not run on hardware
//
#include <hip/hip_runtime.h>
#include <hip/hip_bf16.h>

typedef __attribute__((ext_vector_type(16))) _Float16 v16h;
typedef __attribute__((ext_vector_type(8)))  float    v8f;

#define B_TOT   4096
#define NLAG    20
#define NT      100
#define DIMH    128
#define DT      0.01f
#define MU1     0.06f
#define MU2     0.02f
#define RRATE   0.02f
#define SIGMA   0.2f
#define BETA    0.05f
#define ETA     0.5f
#define LAMBD   1.0f
#define LOGEPS  1e-06f
#define UTIL_PENALTY 100.0f

#define NKC     5            // 130 padded to 5 K-chunks of 32
#define NGATE   512
#define TILE_M  16
#define WFRAG_ELEMS (32 * NKC * 32 * 16)   // nt(32) * kc(5) * lane(32) * halfs(16)

// CDNA5 has a hardware V_TANH_F32 transcendental — use it if the toolchain
// exposes the builtin; fall back to libm otherwise.
#if __has_builtin(__builtin_amdgcn_tanhf)
__device__ __forceinline__ float fast_tanh(float v) { return __builtin_amdgcn_tanhf(v); }
#elif __has_builtin(__builtin_amdgcn_tanh_f32)
__device__ __forceinline__ float fast_tanh(float v) { return __builtin_amdgcn_tanh_f32(v); }
#else
__device__ __forceinline__ float fast_tanh(float v) { return tanhf(v); }
#endif

// sigmoid(x) = 0.5 + 0.5*tanh(x/2): one TRANS op + 2 VALU when HW tanh exists.
__device__ __forceinline__ float sigm(float v) { return 0.5f + 0.5f * fast_tanh(0.5f * v); }

// Write one f16 value of z (the WMMA A-matrix, 16x32 per K-chunk) into the
// fragment layout: lane l<16 holds row M=l, K-halves {0..7,16..23}; lane l+16
// holds K-halves {8..15,24..31}; VGPR v holds K pair (v<4?0:16)+(v%4)*2 (+8 hi).
__device__ __forceinline__ void write_z(_Float16* z, int m, int K, float val) {
    int kc = K >> 5;
    int Kl = K & 31;
    int v  = ((Kl & 16) ? 4 : 0) + ((Kl & 7) >> 1);
    int p  = Kl & 1;
    int li = ((Kl & 8) ? 16 : 0) + m;
    z[(kc * 32 + li) * 16 + 2 * v + p] = (_Float16)val;
}

// ---------------------------------------------------------------------------
// Prep: repack W_all = [W_f | W_i | W_o | W_c] (130x512, f32) into f16
// B-matrix fragments: frag[nt][kc][lane][e] = W_all[K = kc*32+(e/8)*16+(lane/16)*8+e%8]
//                                                  [N = nt*16 + lane%16]
// (K >= 130 zero-padded.)
// ---------------------------------------------------------------------------
__global__ void prep_weights(const float* __restrict__ Wf, const float* __restrict__ Wi,
                             const float* __restrict__ Wo, const float* __restrict__ Wc,
                             _Float16* __restrict__ wfrag) {
    int idx = blockIdx.x * blockDim.x + threadIdx.x;
    if (idx >= WFRAG_ELEMS) return;
    int e    = idx & 15;
    int lane = (idx >> 4) & 31;
    int kc   = (idx >> 9) % NKC;
    int nt   = idx / (NKC * 512);
    int K = kc * 32 + ((e & 8) ? 16 : 0) + ((lane >> 4) << 3) + (e & 7);
    int N = nt * 16 + (lane & 15);
    float v = 0.0f;
    if (K < DIMH + 2) {
        int g = N >> 7, c = N & 127;
        const float* W = (g == 0) ? Wf : (g == 1) ? Wi : (g == 2) ? Wo : Wc;
        v = W[K * DIMH + c];
    }
    wfrag[idx] = (_Float16)v;
}

// ---------------------------------------------------------------------------
// Main recurrent kernel: one block = 16 batch rows, 8 waves, 120 steps.
// Weights become register-resident (compiler hoists the loop-invariant
// B-fragment loads, using CDNA5's >256 VGPR space via s_set_vgpr_msb).
// ---------------------------------------------------------------------------
__global__ void __launch_bounds__(256)
polstm_kernel(const float* __restrict__ dw_sample, const float* __restrict__ x_init,
              const float* __restrict__ b_f, const float* __restrict__ b_i,
              const float* __restrict__ b_o, const float* __restrict__ b_c,
              const float* __restrict__ W_pi, const float* __restrict__ b_pi,
              const _Float16* __restrict__ wfrag, float* __restrict__ out)
{
    __shared__ alignas(32) _Float16 z_lds[NKC * 32 * 16];  // A fragments (5 KB)
    __shared__ float gates_lds[TILE_M * NGATE];            // 32 KB
    __shared__ float h_lds[TILE_M * DIMH];                 // 8 KB
    __shared__ float pi_lds[32];

    const int tid  = threadIdx.x;
    const int lane = tid & 31;
    const int wave = tid >> 5;
    const int m_ew = tid >> 4;            // element-wise: row this thread updates
    const int jb   = (tid & 15) << 3;     // element-wise: 8 hidden units base
    const int blk  = blockIdx.x;
    const int rowg_ew = blk * TILE_M + m_ew;

    // Warm the weight fragments (shared, L2-resident across all blocks).
    for (int off = tid * 64; off < WFRAG_ELEMS; off += 256 * 64)
        __builtin_prefetch(wfrag + off, 0, 1);

    // Per-thread gate biases (8 hidden units x 4 gates).
    float bfv[8], biv[8], bov[8], bcv[8];
    #pragma unroll
    for (int jj = 0; jj < 8; ++jj) {
        bfv[jj] = b_f[jb + jj]; biv[jj] = b_i[jb + jj];
        bov[jj] = b_o[jb + jj]; bcv[jj] = b_c[jb + jj];
    }

    // Zero z (covers K=130..159 padding permanently).
    for (int i = tid; i < NKC * 32 * 16; i += 256) z_lds[i] = (_Float16)0.0f;
    __syncthreads();

    // h0 = [x_init[:,0], 0 ...]; C0 = h0.
    float Cst[8];
    {
        float x0 = x_init[rowg_ew * (NLAG + 1)];
        #pragma unroll
        for (int jj = 0; jj < 8; ++jj) {
            int j = jb + jj;
            float v = (j == 0) ? x0 : 0.0f;
            Cst[jj] = v;
            h_lds[m_ew * DIMH + j] = v;
            write_z(z_lds, m_ew, 1 + j, v);   // h occupies K = 1..128
        }
    }

    const int rowg = blk * TILE_M + tid;      // valid when tid < 16
    float xs = 0.0f, ys = 0.0f, reward = 0.0f;

    // ---- one LSTM cell evaluation (callers wrote x@K=0, t@K=129 into z) ----
    auto lstm_step = [&]() {
        __syncthreads();                      // z ready
        // --- WMMA tile GEMM: (16 x 160) x (160 x 512), this wave: 64 cols ---
        const v16h* zf = reinterpret_cast<const v16h*>(z_lds);
        v16h A[NKC];
        #pragma unroll
        for (int kc = 0; kc < NKC; ++kc) A[kc] = zf[kc * 32 + lane];
        const v16h* wf = reinterpret_cast<const v16h*>(wfrag);
        const int mbase = (lane >> 4) << 3;   // D layout: lanes>=16 hold M=8..15
        #pragma unroll
        for (int nt = 0; nt < 4; ++nt) {
            const int ntg = wave * 4 + nt;
            v8f acc = {};
            #pragma unroll
            for (int kc = 0; kc < NKC; ++kc) {
                v16h bm = wf[(ntg * NKC + kc) * 32 + lane];
                acc = __builtin_amdgcn_wmma_f32_16x16x32_f16(
                          false, A[kc], false, bm, (short)0, acc, false, false);
            }
            const int colb = ntg * 16 + (lane & 15);
            #pragma unroll
            for (int r = 0; r < 8; ++r)
                gates_lds[(mbase + r) * NGATE + colb] = acc[r];
        }
        __syncthreads();                      // gates ready
        // --- gate nonlinearities + state update (8 units per thread) ---
        #pragma unroll
        for (int jj = 0; jj < 8; ++jj) {
            int j = jb + jj;
            float gf = gates_lds[m_ew * NGATE + j        ] + bfv[jj];
            float gi = gates_lds[m_ew * NGATE + 128 + j  ] + biv[jj];
            float go = gates_lds[m_ew * NGATE + 256 + j  ] + bov[jj];
            float gc = gates_lds[m_ew * NGATE + 384 + j  ] + bcv[jj];
            float Cv = Cst[jj] * sigm(gf) + fast_tanh(gc) * sigm(gi);
            Cst[jj] = Cv;
            float hv = sigm(go) * fast_tanh(Cv);
            h_lds[m_ew * DIMH + j] = hv;
            write_z(z_lds, m_ew, 1 + j, hv);
        }
        __syncthreads();                      // h ready
    };

    // ---------------- warm-up: 20 lagged steps ----------------
    for (int k = 0; k < NLAG; ++k) {
        if (tid < TILE_M) {
            float xc = x_init[rowg * (NLAG + 1) + 1 + k];
            write_z(z_lds, tid, 0,   xc);
            write_z(z_lds, tid, 129, (float)(k - NLAG) * DT);
        }
        lstm_step();
    }

    // ---------------- initialize SDE state ----------------
    if (tid < TILE_M) {
        xs = x_init[rowg * (NLAG + 1) + NLAG];
        float acc = 0.0f;
        #pragma unroll
        for (int i = 0; i < NLAG; ++i)
            acc += __expf(-LAMBD * DT * (float)(NLAG - i)) * x_init[rowg * (NLAG + 1) + 1 + i];
        const float geom = (1.0f - __expf(-LAMBD * DT)) / LAMBD * __expf(-LAMBD * DT * (float)NLAG);
        ys = acc * DT + geom * x_init[rowg * (NLAG + 1)];
        reward = 0.0f;
    }

    const float y_decay = __expf(-DT * LAMBD);

    // ---------------- 100 SDE steps ----------------
    for (int t_i = 0; t_i < NT; ++t_i) {
        if (tid < TILE_M) {
            write_z(z_lds, tid, 0,   xs);
            write_z(z_lds, tid, 129, (float)t_i * DT);
        }
        lstm_step();
        // pi = h @ W_pi + b_pi : 16 rows x 2 outputs = 32 dots by wave 0
        if (tid < 32) {
            int m = tid >> 1, k = tid & 1;
            float acc = b_pi[k];
            for (int j = 0; j < DIMH; ++j)
                acc += h_lds[m * DIMH + j] * W_pi[j * 2 + k];
            pi_lds[tid] = acc;
        }
        __syncthreads();
        if (tid < TILE_M) {
            float p0 = 2.0f * sigm(pi_lds[tid * 2 + 0]);
            float p1 = 2.0f * sigm(pi_lds[tid * 2 + 1]);
            float tf = (float)t_i;
            float inst = __logf(fmaxf(p0 * xs, 0.0f) + LOGEPS) * __expf(-BETA * tf * DT)
                       - fmaxf(-xs, 0.0f) * UTIL_PENALTY;
            reward += inst * DT;
            float dw = dw_sample[rowg * NT + t_i];
            float dx = ((MU1 - RRATE) * p1 - p0 + RRATE) * xs + MU2 * ys;
            xs = xs + dx * DT + SIGMA * xs * p1 * dw;
            ys = ys * y_decay + xs * DT;
        }
    }

    // ---------------- terminal reward ----------------
    if (tid < TILE_M) {
        const float final_disc = __expf(-BETA * (float)NT * DT);
        reward += __logf(fmaxf(xs + ETA * ys, 0.0f) + LOGEPS) / BETA * final_disc
                - fmaxf(-xs, 0.0f) * UTIL_PENALTY;
        out[rowg] = -reward;
    }
}

extern "C" void kernel_launch(void* const* d_in, const int* in_sizes, int n_in,
                              void* d_out, int out_size, void* d_ws, size_t ws_size,
                              hipStream_t stream) {
    (void)in_sizes; (void)n_in; (void)out_size; (void)ws_size;
    const float* dw_sample = (const float*)d_in[0];
    const float* x_init    = (const float*)d_in[1];
    const float* W_f = (const float*)d_in[2];  const float* b_f = (const float*)d_in[3];
    const float* W_i = (const float*)d_in[4];  const float* b_i = (const float*)d_in[5];
    const float* W_o = (const float*)d_in[6];  const float* b_o = (const float*)d_in[7];
    const float* W_c = (const float*)d_in[8];  const float* b_c = (const float*)d_in[9];
    const float* W_pi = (const float*)d_in[10]; const float* b_pi = (const float*)d_in[11];

    _Float16* wfrag = (_Float16*)d_ws;   // 160 KB of fragments

    prep_weights<<<(WFRAG_ELEMS + 255) / 256, 256, 0, stream>>>(W_f, W_i, W_o, W_c, wfrag);
    polstm_kernel<<<B_TOT / TILE_M, 256, 0, stream>>>(
        dw_sample, x_init, b_f, b_i, b_o, b_c, W_pi, b_pi, wfrag, (float*)d_out);
}
